// StdAbsBlock_32590211842339
// MI455X (gfx1250) — matrix operand order repaired and checked by reading, not compiled, not run on hardware
//
#include <hip/hip_runtime.h>
#include <cmath>

// ---------------------------------------------------------------------------
// Types for CDNA5 WMMA (wave32): v_wmma_f32_16x16x32_bf16
// ---------------------------------------------------------------------------
typedef __attribute__((ext_vector_type(16))) __bf16 v16bf;
typedef __attribute__((ext_vector_type(8)))  float  v8f;
typedef __attribute__((ext_vector_type(4)))  unsigned int u32x4;
typedef __attribute__((ext_vector_type(8)))  int          i32x8;
typedef __attribute__((ext_vector_type(4)))  int          i32x4;

#if __has_builtin(__builtin_amdgcn_tensor_load_to_lds) && __has_builtin(__builtin_amdgcn_s_wait_tensorcnt)
#define HAVE_TDM 1
#else
#define HAVE_TDM 0
#endif

static __device__ __forceinline__ unsigned short f2bf(float x) {
    unsigned int u = __builtin_bit_cast(unsigned int, x);
    unsigned int r = u + 0x7fffu + ((u >> 16) & 1u);   // round-to-nearest-even
    return (unsigned short)(r >> 16);
}

static __device__ __forceinline__ unsigned int pack2bf(float a, float b) {
    return (unsigned int)f2bf(a) | ((unsigned int)f2bf(b) << 16);
}

union FragU { uint4 u[2]; v16bf b; };

// A-fragment (16x32 bf16, M x K): lane l = row; per half, 16 contiguous K.
// elements e -> K = k0 + hi*16 + e
static __device__ __forceinline__ v16bf load_a_frag(const unsigned short* __restrict__ row,
                                                    int k0, int hi) {
    FragU f;
    f.u[0] = *(const uint4*)(row + k0 + hi * 16);
    f.u[1] = *(const uint4*)(row + k0 + hi * 16 + 8);
    return f.b;
}

// B-fragment (32x16 bf16, K x N): lane l = column; per half, 16 contiguous K.
// elements e -> K = k0 + hi*16 + e  (one contiguous 32B run)
static __device__ __forceinline__ v16bf load_b_frag(const unsigned short* __restrict__ col,
                                                    int k0, int hi) {
    FragU f;
    const uint4* p = (const uint4*)(col + k0 + hi * 16);
    f.u[0] = p[0];
    f.u[1] = p[1];
    return f.b;
}

static __device__ __forceinline__ v8f wmma_bf16(v16bf a, v16bf b, v8f c) {
    return __builtin_amdgcn_wmma_f32_16x16x32_bf16(false, a, false, b,
                                                   (short)0, c, false, false);
}

#if HAVE_TDM
// ---------------------------------------------------------------------------
// TDM: issue one 2D-tile DMA (global -> LDS) via a Tensor DMA Descriptor.
// Tile: tile_h rows x tile_w bf16 elements, row stride = stride (elements).
// D# bit layout per CDNA5 ISA ch.8 (groups 2/3 zero: <=2D tensor).
// clang-23 toolchain: 6-arg builtin (g0, g1, g2, g3, extra, cpol).
// ---------------------------------------------------------------------------
static __device__ __forceinline__ void tdm_load_2d(unsigned lds_off,
                                                   const unsigned short* gptr,
                                                   unsigned tile_w, unsigned tile_h,
                                                   unsigned long long stride) {
    const unsigned long long ga = (unsigned long long)gptr;
    u32x4 g0;
    g0[0] = 1u;                                           // count=1, user D#
    g0[1] = lds_off;                                      // LDS byte address
    g0[2] = (unsigned)(ga & 0xffffffffu);                 // global_addr[31:0]
    g0[3] = (unsigned)((ga >> 32) & 0x01ffffffu)          // global_addr[56:32]
          | 0x80000000u;                                  // type=2 ("image")
    i32x8 g1;
    g1[0] = (int)(1u << 16);                              // data_size=1 (2B)
    g1[1] = (int)((tile_w & 0xffffu) << 16);              // tensor_dim0[15:0]
    g1[2] = (int)((tile_w >> 16) | ((tile_h & 0xffffu) << 16)); // d0[31:16]|d1[15:0]
    g1[3] = (int)((tile_h >> 16) | (tile_w << 16));       // d1[31:16]|tile_dim0
    g1[4] = (int)(tile_h & 0xffffu);                      // tile_dim1 (tile_dim2=0)
    g1[5] = (int)(unsigned)(stride & 0xffffffffu);        // dim0_stride[31:0]
    g1[6] = (int)(unsigned)((stride >> 32) & 0xffffu);    // dim0_stride[47:32]
    g1[7] = 0;
    const i32x4 z4 = {0, 0, 0, 0};
    const i32x8 z8 = {0, 0, 0, 0, 0, 0, 0, 0};
    __builtin_amdgcn_tensor_load_to_lds(g0, g1, z4, z4, z8, 0);
}
#endif

// ---------------------------------------------------------------------------
// fp32 -> bf16 conversion
// ---------------------------------------------------------------------------
__global__ __launch_bounds__(256) void k_f32_to_bf16(const float* __restrict__ in,
                                                     unsigned short* __restrict__ out,
                                                     int n) {
    int i = blockIdx.x * 256 + threadIdx.x;
    if (i < n) out[i] = f2bf(in[i]);
}

// ---------------------------------------------------------------------------
// LayerNorm over rows of 1024, write bf16. One block per row, 256 threads.
// ---------------------------------------------------------------------------
__global__ __launch_bounds__(256) void k_layernorm_bf16(const float* __restrict__ x,
                                                        const float* __restrict__ w,
                                                        const float* __restrict__ b,
                                                        unsigned short* __restrict__ out) {
    __shared__ float red0[8];
    __shared__ float red1[8];
    const int row = blockIdx.x;
    const float* xr = x + (size_t)row * 1024;
    float v[4], s = 0.f, s2 = 0.f;
#pragma unroll
    for (int i = 0; i < 4; ++i) {
        float t = xr[threadIdx.x + i * 256];
        v[i] = t; s += t; s2 += t * t;
    }
#pragma unroll
    for (int m = 1; m < 32; m <<= 1) {
        s  += __shfl_xor(s,  m, 32);
        s2 += __shfl_xor(s2, m, 32);
    }
    const int wave = threadIdx.x >> 5, lane = threadIdx.x & 31;
    if (lane == 0) { red0[wave] = s; red1[wave] = s2; }
    __syncthreads();
    float ts = 0.f, ts2 = 0.f;
#pragma unroll
    for (int i = 0; i < 8; ++i) { ts += red0[i]; ts2 += red1[i]; }
    const float mu  = ts * (1.0f / 1024.0f);
    const float var = ts2 * (1.0f / 1024.0f) - mu * mu;
    const float inv = rsqrtf(var + 1e-5f);
#pragma unroll
    for (int i = 0; i < 4; ++i) {
        int c = threadIdx.x + i * 256;
        out[(size_t)row * 1024 + c] = f2bf((v[i] - mu) * inv * w[c] + b[c]);
    }
}

// ---------------------------------------------------------------------------
// Shared epilogue for the GEMMs.
// ---------------------------------------------------------------------------
template <bool BIAS, bool GELU, bool RESID, bool OUTBF>
static __device__ __forceinline__ void gemm_epilogue(v8f (&acc)[2][4],
                                                     int row0, int col0, int l, int hi,
                                                     const float* __restrict__ bias,
                                                     const float* __restrict__ resid,
                                                     float* __restrict__ Cf,
                                                     unsigned short* __restrict__ Cb,
                                                     int N) {
#pragma unroll
    for (int i = 0; i < 2; ++i) {
#pragma unroll
        for (int j = 0; j < 4; ++j) {
            const int cg = col0 + j * 16 + l;
            const float bv = BIAS ? bias[cg] : 0.0f;
#pragma unroll
            for (int v = 0; v < 8; ++v) {
                const int rg = row0 + i * 16 + v + 8 * hi;
                float val = acc[i][j][v] + bv;
                if (GELU)  val = 0.5f * val * (1.0f + erff(val * 0.70710678118654752f));
                if (RESID) val += resid[(size_t)rg * N + cg];
                if (OUTBF) Cb[(size_t)rg * N + cg] = f2bf(val);
                else       Cf[(size_t)rg * N + cg] = val;
            }
        }
    }
}

#if HAVE_TDM
// ---------------------------------------------------------------------------
// bf16 WMMA GEMM with TDM double-buffered LDS staging:
//   C[M,N] = A[M,K] * B[N,K]^T  (+bias, gelu, residual)
// Block tile 128x128, k-step 32. Wave 0 issues two tensor_load_to_lds
// descriptors (A-tile 128x32, B-tile 128x32) per step and waits TENSORcnt;
// all 8 waves (4x2, 32x64 each) consume fragments from LDS (ds_load_b128).
// ---------------------------------------------------------------------------
template <bool BIAS, bool GELU, bool RESID, bool OUTBF>
__global__ __launch_bounds__(256) void k_gemm_bf16_tdm(const unsigned short* __restrict__ A,
                                                       const unsigned short* __restrict__ B,
                                                       const float* __restrict__ bias,
                                                       const float* __restrict__ resid,
                                                       float* __restrict__ Cf,
                                                       unsigned short* __restrict__ Cb,
                                                       int N, int K) {
    __shared__ unsigned short sA[2][128 * 32];
    __shared__ unsigned short sB[2][128 * 32];

    const int tid  = threadIdx.x;
    const int lane = tid & 31;
    const int wave = tid >> 5;
    const int l    = lane & 15;
    const int hi   = lane >> 4;
    const int wm   = wave >> 1;        // 0..3
    const int wn   = wave & 1;         // 0..1
    const int row0 = blockIdx.y * 128 + wm * 32;
    const int col0 = blockIdx.x * 128 + wn * 64;

    const unsigned short* Ablk = A + (size_t)blockIdx.y * 128 * K;
    const unsigned short* Bblk = B + (size_t)blockIdx.x * 128 * K;
    const unsigned ldsA[2] = { (unsigned)(uintptr_t)(void*)&sA[0][0],
                               (unsigned)(uintptr_t)(void*)&sA[1][0] };
    const unsigned ldsB[2] = { (unsigned)(uintptr_t)(void*)&sB[0][0],
                               (unsigned)(uintptr_t)(void*)&sB[1][0] };

    v8f acc[2][4] = {};
    const int nsteps = K >> 5;

    if (wave == 0) {
        tdm_load_2d(ldsA[0], Ablk, 32, 128, (unsigned long long)K);
        tdm_load_2d(ldsB[0], Bblk, 32, 128, (unsigned long long)K);
    }

    for (int i = 0; i < nsteps; ++i) {
        const int kb = i & 1;
        if (wave == 0) {
            if (i + 1 < nsteps) {                       // prefetch next tile pair
                tdm_load_2d(ldsA[kb ^ 1], Ablk + (size_t)(i + 1) * 32, 32, 128,
                            (unsigned long long)K);
                tdm_load_2d(ldsB[kb ^ 1], Bblk + (size_t)(i + 1) * 32, 32, 128,
                            (unsigned long long)K);
                __builtin_amdgcn_s_wait_tensorcnt(2);   // pair i done, i+1 in flight
            } else {
                __builtin_amdgcn_s_wait_tensorcnt(0);
            }
        }
        __syncthreads();                                // buffer kb ready

        const unsigned short* a0p = &sA[kb][(wm * 32 + l) * 32];
        const unsigned short* a1p = &sA[kb][(wm * 32 + 16 + l) * 32];
        v16bf a0 = load_a_frag(a0p, 0, hi);
        v16bf a1 = load_a_frag(a1p, 0, hi);
#pragma unroll
        for (int j = 0; j < 4; ++j) {
            const unsigned short* bp = &sB[kb][(wn * 64 + j * 16 + l) * 32];
            v16bf bj = load_b_frag(bp, 0, hi);
            acc[0][j] = wmma_bf16(a0, bj, acc[0][j]);
            acc[1][j] = wmma_bf16(a1, bj, acc[1][j]);
        }
        __syncthreads();                                // done reading kb before re-DMA
    }

    gemm_epilogue<BIAS, GELU, RESID, OUTBF>(acc, row0, col0, l, hi, bias, resid, Cf, Cb, N);
}
#else
// ---------------------------------------------------------------------------
// Fallback: direct-from-global bf16 WMMA GEMM (L2-resident operands).
// ---------------------------------------------------------------------------
template <bool BIAS, bool GELU, bool RESID, bool OUTBF>
__global__ __launch_bounds__(256) void k_gemm_bf16_tdm(const unsigned short* __restrict__ A,
                                                       const unsigned short* __restrict__ B,
                                                       const float* __restrict__ bias,
                                                       const float* __restrict__ resid,
                                                       float* __restrict__ Cf,
                                                       unsigned short* __restrict__ Cb,
                                                       int N, int K) {
    const int tid  = threadIdx.x;
    const int lane = tid & 31;
    const int wave = tid >> 5;
    const int l    = lane & 15;
    const int hi   = lane >> 4;
    const int wm   = wave >> 1;
    const int wn   = wave & 1;
    const int row0 = blockIdx.y * 128 + wm * 32;
    const int col0 = blockIdx.x * 128 + wn * 64;

    v8f acc[2][4] = {};
    const unsigned short* Ar0 = A + (size_t)(row0 + l)      * K;
    const unsigned short* Ar1 = A + (size_t)(row0 + 16 + l) * K;
    const unsigned short* Bc[4] = {
        B + (size_t)(col0 + l)      * K, B + (size_t)(col0 + 16 + l) * K,
        B + (size_t)(col0 + 32 + l) * K, B + (size_t)(col0 + 48 + l) * K
    };
    for (int k = 0; k < K; k += 32) {
        __builtin_prefetch(Ar0 + k + 128, 0, 3);
        __builtin_prefetch(Ar1 + k + 128, 0, 3);
        v16bf a0 = load_a_frag(Ar0, k, hi);
        v16bf a1 = load_a_frag(Ar1, k, hi);
#pragma unroll
        for (int j = 0; j < 4; ++j) {
            v16bf bj = load_b_frag(Bc[j], k, hi);
            acc[0][j] = wmma_bf16(a0, bj, acc[0][j]);
            acc[1][j] = wmma_bf16(a1, bj, acc[1][j]);
        }
    }
    gemm_epilogue<BIAS, GELU, RESID, OUTBF>(acc, row0, col0, l, hi, bias, resid, Cf, Cb, N);
}
#endif

// ---------------------------------------------------------------------------
// Build V^T: vT[b,h,dh,kv] = qkv_bf[b,kv, 2048 + h*64 + dh]
// ---------------------------------------------------------------------------
__global__ __launch_bounds__(256) void k_build_vT(const unsigned short* __restrict__ qkv,
                                                  unsigned short* __restrict__ vT) {
    const size_t i = (size_t)blockIdx.x * 256 + threadIdx.x;  // 2*16*64*2048 total
    const int kv = (int)(i & 2047);
    const size_t r = i >> 11;
    const int dh = (int)(r & 63);
    const int h  = (int)((r >> 6) & 15);
    const int b  = (int)(r >> 10);
    vT[i] = qkv[((size_t)(b * 2048 + kv)) * 3072 + 2048 + h * 64 + dh];
}

// ---------------------------------------------------------------------------
// Causal flash attention, transposed-score formulation (register-only P).
//   S^T = K · Q^T   (C-layout: row = kv, col = q  -> per-lane softmax stats)
//   O^T = V^T · P   (C-layout: row = dh, col = q  -> per-lane rescale/norm)
// ---------------------------------------------------------------------------
__global__ __launch_bounds__(256) void k_attn(const unsigned short* __restrict__ qkv,
                                              const unsigned short* __restrict__ vT,
                                              unsigned short* __restrict__ out) {
    const int tid  = threadIdx.x;
    const int lane = tid & 31;
    const int wave = tid >> 5;
    const int l    = lane & 15;
    const int hi   = lane >> 4;
    const int bh   = blockIdx.x >> 4;   // b*16 + h
    const int qt   = blockIdx.x & 15;   // q super-tile of 128
    const int b    = bh >> 4;
    const int h    = bh & 15;
    const int q0   = qt * 128 + wave * 16;
    const int q    = q0 + l;            // this lane's query token

    const unsigned short* qrow = qkv + ((size_t)(b * 2048 + q)) * 3072 + h * 64;
    const v16bf qb0 = load_b_frag(qrow, 0, hi);
    const v16bf qb1 = load_b_frag(qrow, 32, hi);

    v8f ov[4] = {};                     // O^T accumulators (dh rows x q cols)
    float m = -__builtin_inff();
    float lsum = 0.0f;

    const int kv_end = q0 + 16;         // causal frontier (exclusive)
    for (int kvc = 0; kvc < kv_end; kvc += 32) {
        v8f st[2];
#pragma unroll
        for (int t = 0; t < 2; ++t) {
            const unsigned short* krow =
                qkv + ((size_t)(b * 2048 + kvc + t * 16 + l)) * 3072 + 1024 + h * 64;
            v16bf ka0 = load_a_frag(krow, 0, hi);
            v16bf ka1 = load_a_frag(krow, 32, hi);
            v8f z = {};
            z     = wmma_bf16(ka0, qb0, z);
            st[t] = wmma_bf16(ka1, qb1, z);
        }

        const bool full = (kvc + 31 <= q0);   // wave-uniform: chunk unmasked
        float p0[8], p1[8];
        float cm = -__builtin_inff();
#pragma unroll
        for (int v = 0; v < 8; ++v) {
            float s0 = st[0][v] * 0.125f;
            float s1 = st[1][v] * 0.125f;
            if (!full) {
                const int kv0 = kvc + v + 8 * hi;
                if (kv0      > q) s0 = -__builtin_inff();
                if (kv0 + 16 > q) s1 = -__builtin_inff();
            }
            p0[v] = s0; p1[v] = s1;
            cm = fmaxf(cm, fmaxf(s0, s1));
        }
        cm = fmaxf(cm, __shfl_xor(cm, 16, 32));      // combine the two halves
        const float mn = fmaxf(m, cm);
        const float f  = __expf(m - mn);
        float rs = 0.0f;
#pragma unroll
        for (int v = 0; v < 8; ++v) {
            p0[v] = __expf(p0[v] - mn);
            p1[v] = __expf(p1[v] - mn);
            rs += p0[v] + p1[v];
        }
        rs += __shfl_xor(rs, 16, 32);
        lsum = lsum * f + rs;
        m = mn;
#pragma unroll
        for (int nn = 0; nn < 4; ++nn)
#pragma unroll
            for (int v = 0; v < 8; ++v) ov[nn][v] *= f;

        // ---- build P as a B-fragment: one packed cross-half swap -----------
        unsigned int p0p[4], p1p[4], oth[4];
#pragma unroll
        for (int j = 0; j < 4; ++j) {
            p0p[j] = pack2bf(p0[2 * j], p0[2 * j + 1]);
            p1p[j] = pack2bf(p1[2 * j], p1[2 * j + 1]);
        }
#pragma unroll
        for (int j = 0; j < 4; ++j) {
            const unsigned int send = hi ? p0p[j] : p1p[j];
            oth[j] = (unsigned int)__shfl_xor((int)send, 16, 32);
        }
        FragU pf;
        pf.u[0] = make_uint4(hi ? oth[0] : p0p[0], hi ? oth[1] : p0p[1],
                             hi ? oth[2] : p0p[2], hi ? oth[3] : p0p[3]);
        pf.u[1] = make_uint4(hi ? p1p[0] : oth[0], hi ? p1p[1] : oth[1],
                             hi ? p1p[2] : oth[2], hi ? p1p[3] : oth[3]);

        // ---- O^T += V^T * P  (A = V rows over dh, K = kv chunk) ------------
#pragma unroll
        for (int nn = 0; nn < 4; ++nn) {
            const unsigned short* vrow = vT + ((size_t)(bh * 64 + nn * 16 + l)) * 2048;
            v16bf va = load_a_frag(vrow, kvc, hi);
            ov[nn] = wmma_bf16(va, pf.b, ov[nn]);
        }
    }

    const float inv = 1.0f / lsum;
    unsigned short* orow = out + ((size_t)(b * 2048 + q)) * 1024 + h * 64;
#pragma unroll
    for (int nn = 0; nn < 4; ++nn) {
        uint4 pkt;
        pkt.x = pack2bf(ov[nn][0] * inv, ov[nn][1] * inv);
        pkt.y = pack2bf(ov[nn][2] * inv, ov[nn][3] * inv);
        pkt.z = pack2bf(ov[nn][4] * inv, ov[nn][5] * inv);
        pkt.w = pack2bf(ov[nn][6] * inv, ov[nn][7] * inv);
        *(uint4*)(orow + nn * 16 + 8 * hi) = pkt;   // dh = nn*16 + 8*hi + 0..7
    }
}

// ---------------------------------------------------------------------------
// Host driver: full transformer block pipeline on `stream`.
// ---------------------------------------------------------------------------
extern "C" void kernel_launch(void* const* d_in, const int* in_sizes, int n_in,
                              void* d_out, int out_size, void* d_ws, size_t ws_size,
                              hipStream_t stream) {
    (void)in_sizes; (void)n_in; (void)out_size; (void)ws_size;

    const float* x      = (const float*)d_in[0];   // (2,2048,1024)
    const float* ln1_w  = (const float*)d_in[1];
    const float* ln1_b  = (const float*)d_in[2];
    const float* ln2_w  = (const float*)d_in[3];
    const float* ln2_b  = (const float*)d_in[4];
    const float* qkv_w  = (const float*)d_in[5];   // (3072,1024)
    const float* o_w    = (const float*)d_in[6];   // (1024,1024)
    const float* ffn_w1 = (const float*)d_in[7];   // (4096,1024)
    const float* ffn_b1 = (const float*)d_in[8];
    const float* ffn_w2 = (const float*)d_in[9];   // (1024,4096)
    const float* ffn_b2 = (const float*)d_in[10];
    float* outp = (float*)d_out;

    const int M = 4096;                 // tokens (2*2048)

    char* ws = (char*)d_ws;
    size_t off = 0;
    auto alloc = [&](size_t bytes) { char* p = ws + off; off += (bytes + 255) & ~(size_t)255; return p; };
    unsigned short* hbf    = (unsigned short*)alloc((size_t)M * 1024 * 2);
    unsigned short* wqkv   = (unsigned short*)alloc((size_t)3072 * 1024 * 2);
    unsigned short* wo     = (unsigned short*)alloc((size_t)1024 * 1024 * 2);
    unsigned short* wf1    = (unsigned short*)alloc((size_t)4096 * 1024 * 2);
    unsigned short* wf2    = (unsigned short*)alloc((size_t)1024 * 4096 * 2);
    unsigned short* qkvbf  = (unsigned short*)alloc((size_t)M * 3072 * 2);
    unsigned short* vT     = (unsigned short*)alloc((size_t)2 * 16 * 64 * 2048 * 2);
    unsigned short* attnbf = (unsigned short*)alloc((size_t)M * 1024 * 2);
    float*          x1     = (float*)alloc((size_t)M * 1024 * 4);
    unsigned short* h2bf   = (unsigned short*)alloc((size_t)M * 1024 * 2);
    unsigned short* ff1bf  = (unsigned short*)alloc((size_t)M * 4096 * 2);

    auto cvt = [&](const float* src, unsigned short* dst, int n) {
        k_f32_to_bf16<<<(n + 255) / 256, 256, 0, stream>>>(src, dst, n);
    };
    cvt(qkv_w,  wqkv, 3072 * 1024);
    cvt(o_w,    wo,   1024 * 1024);
    cvt(ffn_w1, wf1,  4096 * 1024);
    cvt(ffn_w2, wf2,  1024 * 4096);

    k_layernorm_bf16<<<M, 256, 0, stream>>>(x, ln1_w, ln1_b, hbf);

    k_gemm_bf16_tdm<false, false, false, true><<<dim3(3072 / 128, M / 128), 256, 0, stream>>>(
        hbf, wqkv, nullptr, nullptr, nullptr, qkvbf, 3072, 1024);

    k_build_vT<<<(2 * 16 * 64 * 2048) / 256, 256, 0, stream>>>(qkvbf, vT);

    k_attn<<<512, 256, 0, stream>>>(qkvbf, vT, attnbf);

    k_gemm_bf16_tdm<false, false, true, false><<<dim3(1024 / 128, M / 128), 256, 0, stream>>>(
        attnbf, wo, nullptr, x, x1, nullptr, 1024, 1024);

    k_layernorm_bf16<<<M, 256, 0, stream>>>(x1, ln2_w, ln2_b, h2bf);

    k_gemm_bf16_tdm<true, true, false, true><<<dim3(4096 / 128, M / 128), 256, 0, stream>>>(
        h2bf, wf1, ffn_b1, nullptr, nullptr, ff1bf, 4096, 1024);

    k_gemm_bf16_tdm<true, false, true, false><<<dim3(1024 / 128, M / 128), 256, 0, stream>>>(
        ff1bf, wf2, ffn_b2, x1, outp, nullptr, 1024, 4096);
}
